// CustomRNN_47751446397338
// MI455X (gfx1250) — 16-bit and 8-bit matrix operand order repaired and checked by reading, compiled, not run on hardware
//
#include <hip/hip_runtime.h>

#define B_  8
#define T_  4096
#define I_  128
#define H_  256
#define G3  768
#define H2_ 512
#define O_  64

typedef __attribute__((ext_vector_type(16))) __bf16        v16bf;
typedef __attribute__((ext_vector_type(8)))  float         v8f;
typedef __attribute__((ext_vector_type(8)))  int           v8i;
typedef __attribute__((ext_vector_type(4)))  int           v4i;
typedef __attribute__((ext_vector_type(4)))  unsigned int  v4u;

#if __has_builtin(__builtin_amdgcn_tensor_load_to_lds)
#define HAVE_TDM 1
#else
#define HAVE_TDM 0
#endif

__device__ __forceinline__ float sigmoidf_(float x){ return 1.0f/(1.0f+__expf(-x)); }

__device__ __forceinline__ v8f wmma_bf16(v16bf a, v16bf b, v8f c){
  return __builtin_amdgcn_wmma_f32_16x16x32_bf16(false, a, false, b, (short)0, c, false, false);
}
__device__ __forceinline__ v8f wmma_fp8(v8i a, v8i b, v8f c){
  return __builtin_amdgcn_wmma_f32_16x16x64_fp8_fp8(a, b, (short)0, c, false, false);
}

__device__ __forceinline__ unsigned char f2e4m3_sw(float x){
  union { float f; unsigned u; } v; v.f = x;
  unsigned s = (v.u >> 24) & 0x80u;
  float a = fabsf(x);
  if (a < 0.015625f){
    int m = (int)rintf(a * 512.0f);
    return (unsigned char)(s | (unsigned)m);
  }
  if (a >= 448.0f) return (unsigned char)(s | 0x7E);
  union { float f; unsigned u; } w; w.f = a;
  unsigned u = w.u;
  unsigned lsb = (u >> 20) & 1u;
  u += 0x7FFFFu + lsb;
  int E = (int)((u >> 23) & 0xFF) - 127 + 7;
  unsigned m3 = (u >> 20) & 7u;
  if (E >= 16) return (unsigned char)(s | 0x7E);
  return (unsigned char)(s | ((unsigned)E << 3) | m3);
}

__device__ __forceinline__ unsigned char f2e4m3(float x){
#if __has_builtin(__builtin_amdgcn_cvt_pk_fp8_f32)
  return (unsigned char)(__builtin_amdgcn_cvt_pk_fp8_f32(x, 0.0f, 0, false) & 0xFF);
#else
  return f2e4m3_sw(x);
#endif
}

__device__ __forceinline__ v16bf load_a_frag(const __bf16* base, int ld, int row0, int k0){
  int lane = threadIdx.x & 31;
  int half = lane >> 4, m = lane & 15;
  const __bf16* p = base + (size_t)(row0 + m) * ld + k0 + 16*half;
  v16bf a;
  #pragma unroll
  for (int v = 0; v < 8; ++v){
    int kb = 2*v;
    a[2*v]   = p[kb];
    a[2*v+1] = p[kb+1];
  }
  return a;
}

__device__ __forceinline__ v16bf load_b_frag_nk(const __bf16* wmat, int ldk, int k0, int n0){
  int lane = threadIdx.x & 31;
  int half = lane >> 4, n = lane & 15;
  const __bf16* p = wmat + (size_t)(n0 + n) * ldk + k0 + 16*half;
  v16bf b;
  #pragma unroll
  for (int e = 0; e < 16; ++e) b[e] = p[e];
  return b;
}

__device__ __forceinline__ v8i load_a_frag_fp8(const unsigned char* base, int ld, int k0){
  int lane = threadIdx.x & 31;
  int half = lane >> 4, m = lane & 15;
  const unsigned char* p = base + (size_t)m * ld + k0 + 16*half;
  v8i a;
  #pragma unroll
  for (int v = 0; v < 8; ++v){
    int kb = ((v >> 2) * 32) + ((v & 3) * 4);
    a[v] = *(const int*)(p + kb);
  }
  return a;
}

__device__ __forceinline__ v8i load_b_frag_fp8_nk(const unsigned char* w, int ldk, int k0, int n0){
  int lane = threadIdx.x & 31;
  int half = lane >> 4, n = lane & 15;
  const unsigned char* p = w + (size_t)(n0 + n) * ldk + k0 + 16*half;
  v8i b;
  #pragma unroll
  for (int v = 0; v < 4; ++v) b[v]     = *(const int*)(p + 4*v);
  #pragma unroll
  for (int v = 0; v < 4; ++v) b[4 + v] = *(const int*)(p + 32 + 4*v);
  return b;
}

__global__ void cvt_f32_bf16(const float* __restrict__ s, __bf16* __restrict__ d, int n){
  int i = blockIdx.x*blockDim.x + threadIdx.x;
  if (i < n) d[i] = (__bf16)s[i];
}
__global__ void cvt_f32_e4m3(const float* __restrict__ s, unsigned char* __restrict__ d, int n){
  int i = blockIdx.x*blockDim.x + threadIdx.x;
  if (i < n) d[i] = f2e4m3(s[i]);
}
__global__ void cvt_transpose512(const float* __restrict__ s, __bf16* __restrict__ d){
  int i = blockIdx.x*blockDim.x + threadIdx.x;
  int k = i >> 9, n = i & 511;
  d[(size_t)n*H2_ + k] = (__bf16)s[(size_t)k*H2_ + n];
}

__global__ __launch_bounds__(256) void gx_gemm(
    const __bf16* __restrict__ xb,
    const __bf16* __restrict__ wih_f, const __bf16* __restrict__ wih_b,
    const float*  __restrict__ bih_f, const float*  __restrict__ bih_b,
    __bf16* __restrict__ gx_f, __bf16* __restrict__ gx_b){
  const int dir = blockIdx.z;
  const __bf16* wih = dir ? wih_b : wih_f;
  const float*  bih = dir ? bih_b : bih_f;
  __bf16* gx = dir ? gx_b : gx_f;

  const int wave = threadIdx.x >> 5;
  const int lane = threadIdx.x & 31;
  const int m0 = (blockIdx.x*8 + wave)*16;
  const int n0 = blockIdx.y*16;

  v8f acc = {};
  #pragma unroll
  for (int kt = 0; kt < 4; ++kt){
    v16bf a = load_a_frag(xb, I_, m0, kt*32);
    v16bf b = load_b_frag_nk(wih, I_, kt*32, n0);
    acc = wmma_bf16(a, b, acc);
  }
  const int half = lane >> 4, nl = lane & 15;
  const float bias = bih[n0 + nl];
  #pragma unroll
  for (int r = 0; r < 8; ++r){
    int m = m0 + r + 8*half;
    gx[(size_t)m*G3 + n0 + nl] = (__bf16)(acc[r] + bias);
  }
}

__global__ __launch_bounds__(512) void gru_scan(
    const unsigned char* __restrict__ whh_f8, const unsigned char* __restrict__ whh_b8,
    const float* __restrict__ bhh_f, const float* __restrict__ bhh_b,
    const __bf16* __restrict__ gx_f, const __bf16* __restrict__ gx_b,
    __bf16* __restrict__ pred){
  const int dir = blockIdx.x;
  const unsigned char* whh = dir ? whh_b8 : whh_f8;
  const float*  bhh = dir ? bhh_b : bhh_f;
  const __bf16* gx  = dir ? gx_b  : gx_f;

  __shared__ unsigned char h8[2][16*H_];
  __shared__ __bf16 gxs[2][8*G3];

  for (int i = threadIdx.x; i < 2*16*H_; i += 512) (&h8[0][0])[i] = 0;

  const int wave = threadIdx.x >> 5;
  const int lane = threadIdx.x & 31;
  const int half = lane >> 4;
  const int nl   = lane & 15;
  const int nw   = wave*16;
  const int j    = nw + nl;
  const float br = bhh[j], bz = bhh[H_ + j], bn = bhh[2*H_ + j];

  v8i wr[4], wz[4], wn[4];
  #pragma unroll
  for (int kt = 0; kt < 4; ++kt){
    wr[kt] = load_b_frag_fp8_nk(whh, H_, kt*64, 0*H_ + nw);
    wz[kt] = load_b_frag_fp8_nk(whh, H_, kt*64, 1*H_ + nw);
    wn[kt] = load_b_frag_fp8_nk(whh, H_, kt*64, 2*H_ + nw);
  }

  auto prefetch = [&](int tt, int buf){
#if HAVE_TDM
    if (wave == 0){
      unsigned lds0 = (unsigned)(size_t)(&gxs[buf][0]);
      unsigned long long ga = (unsigned long long)(size_t)(gx + (size_t)tt * G3);
      v4u g0;
      g0[0] = 1u;
      g0[1] = lds0;
      g0[2] = (unsigned)(ga & 0xFFFFFFFFu);
      g0[3] = (unsigned)((ga >> 32) & 0x01FFFFFFu) | 0x80000000u;
      v8i g1;
      g1[0] = 0x00010000;
      g1[1] = (int)(768u << 16);
      g1[2] = (int)0x80000000u;
      g1[3] = (int)(768u << 16);
      g1[4] = 8;
      g1[5] = T_ * G3;
      g1[6] = 0;
      g1[7] = 0;
      v4i z4 = {0,0,0,0};
      v8i z8 = {0,0,0,0,0,0,0,0};
      __builtin_amdgcn_tensor_load_to_lds(g0, g1, z4, z4, z8, 0);
    }
#else
    unsigned lds0 = (unsigned)(size_t)(&gxs[buf][0]);
    for (int c = threadIdx.x; c < 8*(G3/8); c += 512){
      int r   = c / (G3/8);
      int off = (c % (G3/8)) * 8;
      const __bf16* g = gx + ((size_t)r*T_ + tt)*G3 + off;
      unsigned l = lds0 + (unsigned)(c * 16);
      asm volatile("global_load_async_to_lds_b128 %0, %1, off" :: "v"(l), "v"(g) : "memory");
    }
#endif
  };
  auto prefetch_wait = [&](){
#if HAVE_TDM
    if (wave == 0){
#if __has_builtin(__builtin_amdgcn_s_wait_tensorcnt)
      __builtin_amdgcn_s_wait_tensorcnt(0);
#else
      asm volatile("s_wait_tensorcnt 0x0" ::: "memory");
#endif
    }
#else
    asm volatile("s_wait_asynccnt 0x0" ::: "memory");
#endif
  };

  prefetch(dir ? (T_-1) : 0, 0);
  prefetch_wait();
  __syncthreads();

  float hreg[8];
  #pragma unroll
  for (int r = 0; r < 8; ++r) hreg[r] = 0.0f;

  for (int t = 0; t < T_; ++t){
    const int tt  = dir ? (T_-1 - t) : t;
    const int buf = t & 1;
    if (t + 1 < T_) prefetch(dir ? (T_-2 - t) : (t+1), buf ^ 1);

    const unsigned char* hcur = &h8[buf][0];
    unsigned char*       hnxt = &h8[buf ^ 1][0];

    v8i af[4];
    #pragma unroll
    for (int kt = 0; kt < 4; ++kt) af[kt] = load_a_frag_fp8(hcur, H_, kt*64);

    v8f ar = {}, az = {}, an = {};
    #pragma unroll
    for (int kt = 0; kt < 4; ++kt){
      ar = wmma_fp8(af[kt], wr[kt], ar);
      az = wmma_fp8(af[kt], wz[kt], az);
      an = wmma_fp8(af[kt], wn[kt], an);
    }

    if (half == 0){
      #pragma unroll
      for (int r = 0; r < 8; ++r){
        float xr = (float)gxs[buf][r*G3 + j];
        float xz = (float)gxs[buf][r*G3 + H_  + j];
        float xn = (float)gxs[buf][r*G3 + 2*H_ + j];
        float rg = sigmoidf_(xr + ar[r] + br);
        float zg = sigmoidf_(xz + az[r] + bz);
        float ng = tanhf(xn + rg*(an[r] + bn));
        float hv = (1.0f - zg)*ng + zg*hreg[r];
        hreg[r] = hv;
        hnxt[r*H_ + j] = f2e4m3(hv);
        pred[((size_t)r*T_ + tt)*H2_ + dir*H_ + j] = (__bf16)hv;
      }
    }
    prefetch_wait();
    __syncthreads();
  }
}

__global__ __launch_bounds__(256) void att_scores(
    const __bf16* __restrict__ pred, const __bf16* __restrict__ wattT,
    const float* __restrict__ vatt, float* __restrict__ scores){
  __shared__ float sc[16];
  const int wave = threadIdx.x >> 5, lane = threadIdx.x & 31;
  const int half = lane >> 4, nl = lane & 15;
  const int m0 = blockIdx.x * 16;
  if (threadIdx.x < 16) sc[threadIdx.x] = 0.0f;
  __syncthreads();

  v16bf af[16];
  #pragma unroll
  for (int kt = 0; kt < 16; ++kt) af[kt] = load_a_frag(pred, H2_, m0, kt*32);

  float part[8];
  #pragma unroll
  for (int r = 0; r < 8; ++r) part[r] = 0.0f;

  for (int i = 0; i < 4; ++i){
    const int n0 = (i*8 + wave)*16;
    v8f acc = {};
    #pragma unroll
    for (int kt = 0; kt < 16; ++kt)
      acc = wmma_bf16(af[kt], load_b_frag_nk(wattT, H2_, kt*32, n0), acc);
    const float v = vatt[n0 + nl];
    #pragma unroll
    for (int r = 0; r < 8; ++r) part[r] += tanhf(acc[r]) * v;
  }
  #pragma unroll
  for (int r = 0; r < 8; ++r) atomicAdd(&sc[r + 8*half], part[r]);
  __syncthreads();
  if (threadIdx.x < 16) scores[m0 + threadIdx.x] = sc[threadIdx.x];
}

__global__ __launch_bounds__(256) void softmax_ctx(
    const float* __restrict__ scores, const __bf16* __restrict__ pred,
    float* __restrict__ ctx){
  const int b = blockIdx.x, tid = threadIdx.x;
  __shared__ float red[256];
  const float* s = scores + (size_t)b*T_;

  float mx = -1e30f;
  for (int t = tid; t < T_; t += 256) mx = fmaxf(mx, s[t]);
  red[tid] = mx; __syncthreads();
  for (int o = 128; o > 0; o >>= 1){ if (tid < o) red[tid] = fmaxf(red[tid], red[tid+o]); __syncthreads(); }
  mx = red[0]; __syncthreads();

  float sm = 0.0f;
  for (int t = tid; t < T_; t += 256) sm += __expf(s[t]-mx);
  red[tid] = sm; __syncthreads();
  for (int o = 128; o > 0; o >>= 1){ if (tid < o) red[tid] += red[tid+o]; __syncthreads(); }
  const float inv = 1.0f / red[0];

  const int c0 = tid, c1 = tid + 256;
  float a0 = 0.0f, a1 = 0.0f;
  const __bf16* p = pred + (size_t)b*T_*H2_;
  for (int t = 0; t < T_; ++t){
    float a = __expf(s[t]-mx)*inv;
    a0 += a * (float)p[(size_t)t*H2_ + c0];
    a1 += a * (float)p[(size_t)t*H2_ + c1];
  }
  ctx[(size_t)b*H2_ + c0] = a0;
  ctx[(size_t)b*H2_ + c1] = a1;
}

__global__ __launch_bounds__(64) void head_kernel(
    const float* __restrict__ ctx, const float* __restrict__ wlin,
    const float* __restrict__ blin, float* __restrict__ out){
  const int b = blockIdx.x, o = threadIdx.x;
  __shared__ float red[64];
  __shared__ float smax, ssum;
  float acc = blin[o];
  const float* c = ctx  + (size_t)b*H2_;
  const float* w = wlin + (size_t)o*H2_;
  for (int h = 0; h < H2_; ++h) acc += c[h]*w[h];
  red[o] = acc; __syncthreads();
  if (o == 0){
    float m = red[0];
    for (int i = 1; i < O_; ++i) m = fmaxf(m, red[i]);
    float su = 0.0f;
    for (int i = 0; i < O_; ++i) su += __expf(red[i]-m);
    smax = m; ssum = su;
  }
  __syncthreads();
  out[(size_t)b*O_ + o] = __expf(acc - smax)/ssum;
}

extern "C" void kernel_launch(void* const* d_in, const int* in_sizes, int n_in,
                              void* d_out, int out_size, void* d_ws, size_t ws_size,
                              hipStream_t stream){
  (void)in_sizes; (void)n_in; (void)out_size; (void)ws_size;
  const float* x    = (const float*)d_in[0];
  const float* wihf = (const float*)d_in[1];
  const float* whhf = (const float*)d_in[2];
  const float* bihf = (const float*)d_in[3];
  const float* bhhf = (const float*)d_in[4];
  const float* wihb = (const float*)d_in[5];
  const float* whhb = (const float*)d_in[6];
  const float* bihb = (const float*)d_in[7];
  const float* bhhb = (const float*)d_in[8];
  const float* watt = (const float*)d_in[9];
  const float* vatt = (const float*)d_in[10];
  const float* wlin = (const float*)d_in[11];
  const float* blin = (const float*)d_in[12];
  float* out = (float*)d_out;

  char* wp = (char*)d_ws;
  auto alloc = [&](size_t bytes)->void*{
    void* p = (void*)wp; wp += (bytes + 255) & ~(size_t)255; return p;
  };
  __bf16* xb    = (__bf16*)alloc((size_t)B_*T_*I_*2);
  __bf16* wihfb = (__bf16*)alloc((size_t)G3*I_*2);
  __bf16* wihbb = (__bf16*)alloc((size_t)G3*I_*2);
  unsigned char* whhf8 = (unsigned char*)alloc((size_t)G3*H_);
  unsigned char* whhb8 = (unsigned char*)alloc((size_t)G3*H_);
  __bf16* wattT = (__bf16*)alloc((size_t)H2_*H2_*2);
  __bf16* gxf   = (__bf16*)alloc((size_t)B_*T_*G3*2);
  __bf16* gxb   = (__bf16*)alloc((size_t)B_*T_*G3*2);
  __bf16* pred  = (__bf16*)alloc((size_t)B_*T_*H2_*2);
  float*  scores= (float*) alloc((size_t)B_*T_*4);
  float*  ctx   = (float*) alloc((size_t)B_*H2_*4);

  cvt_f32_bf16<<<dim3((B_*T_*I_+255)/256), dim3(256), 0, stream>>>(x, xb, B_*T_*I_);
  cvt_f32_bf16<<<dim3((G3*I_+255)/256), dim3(256), 0, stream>>>(wihf, wihfb, G3*I_);
  cvt_f32_bf16<<<dim3((G3*I_+255)/256), dim3(256), 0, stream>>>(wihb, wihbb, G3*I_);
  cvt_f32_e4m3<<<dim3((G3*H_+255)/256), dim3(256), 0, stream>>>(whhf, whhf8, G3*H_);
  cvt_f32_e4m3<<<dim3((G3*H_+255)/256), dim3(256), 0, stream>>>(whhb, whhb8, G3*H_);
  cvt_transpose512<<<dim3((H2_*H2_+255)/256), dim3(256), 0, stream>>>(watt, wattT);

  gx_gemm<<<dim3(256, 48, 2), dim3(256), 0, stream>>>(xb, wihfb, wihbb, bihf, bihb, gxf, gxb);

  gru_scan<<<dim3(2), dim3(512), 0, stream>>>(whhf8, whhb8, bhhf, bhhb, gxf, gxb, pred);

  att_scores<<<dim3((B_*T_)/16), dim3(256), 0, stream>>>(pred, wattT, vatt, scores);

  softmax_ctx<<<dim3(B_), dim3(256), 0, stream>>>(scores, pred, ctx);

  head_kernel<<<dim3(B_), dim3(64), 0, stream>>>(ctx, wlin, blin, out);
}
